// GATLayer_52871047414197
// MI455X (gfx1250) — matrix operand order repaired and checked by reading, not compiled, not run on hardware
//
#include <hip/hip_runtime.h>
#include <hip/hip_bf16.h>

#define ALPHA 0.2f

typedef __bf16 bf16;
typedef __attribute__((ext_vector_type(16))) __bf16 v16bf;
typedef __attribute__((ext_vector_type(8)))  float  v8f;

// ---------------------------------------------------------------------------
// Pack W (256 x 128, f32 row-major) into per-lane bf16 WMMA B-fragments.
// Layout: Wp[((kc*8 + nt)*32 + lane)*16 + i]
//   lane 0-15  : col = nt*16 + lane,     K = kc*32 + {0..7, 16..23}[i]
//   lane 16-31 : col = nt*16 + lane-16,  K = kc*32 + {8..15, 24..31}[i]
// ---------------------------------------------------------------------------
__global__ __launch_bounds__(256) void pack_w(const float* __restrict__ W,
                                              unsigned short* __restrict__ Wp_) {
    bf16* Wp = (bf16*)Wp_;
    int idx = blockIdx.x * 256 + threadIdx.x;      // 0 .. 65535
    int i    = idx & 15;
    int lane = (idx >> 4) & 31;
    int nt   = (idx >> 9) & 7;
    int kc   = idx >> 12;
    int k    = kc * 32 + ((lane >> 4) * 8) + (i & 7) + ((i >> 3) * 16);
    int col  = nt * 16 + (lane & 15);
    Wp[idx] = (bf16)W[(size_t)k * 128 + col];
}

// ---------------------------------------------------------------------------
// GEMM: Y(rows x 128) = X(rows x 256) * W, bf16 WMMA, f32 accumulate.
// Block = 256 threads = 8 waves; wave w owns rows [base+w*16, base+w*16+16)
// and all 128 output columns (8 x 16-wide tiles -> 8 v8f accumulators).
// Packed W (64 KB) staged in LDS once per block; A stream software-pipelined
// one K-step ahead; B fragments double-buffered from LDS. rows % 16 == 0.
// ---------------------------------------------------------------------------
__global__ __launch_bounds__(256) void gat_gemm(const float* __restrict__ X,
                                                const unsigned short* __restrict__ Wp_,
                                                float* __restrict__ Y, int rows) {
    __shared__ __align__(32) unsigned short ldsW[32768];   // 64 KB packed W

    // cooperative copy: 4096 uint4 (16 B) moves, 16 per thread
    {
        const uint4* g = (const uint4*)Wp_;
        uint4* l = (uint4*)ldsW;
        int t = threadIdx.x;
#pragma unroll
        for (int i = 0; i < 16; ++i)
            l[t + i * 256] = g[t + i * 256];
    }
    __syncthreads();

    const bf16* Wl = (const bf16*)ldsW;
    const int lane  = threadIdx.x & 31;
    const int wave  = threadIdx.x >> 5;
    const int rowBase = blockIdx.x * 128 + wave * 16;
    if (rowBase >= rows) return;          // wave-uniform; no barriers after this

    const int rlo   = lane & 15;
    const int khalf = lane >> 4;          // 0 or 1
    const float* arow = X + (size_t)(rowBase + rlo) * 256 + khalf * 8;

    v8f acc[8];
#pragma unroll
    for (int nt = 0; nt < 8; ++nt)
        acc[nt] = (v8f){0.f, 0.f, 0.f, 0.f, 0.f, 0.f, 0.f, 0.f};

    // current A fragment (f32), K = kbase + {0..7, 16..23}
    float4 c0 = *(const float4*)(arow);
    float4 c1 = *(const float4*)(arow + 4);
    float4 c2 = *(const float4*)(arow + 16);
    float4 c3 = *(const float4*)(arow + 20);

#pragma unroll
    for (int kc = 0; kc < 8; ++kc) {
        // prefetch next K-step's A while this step's WMMAs run
        float4 n0 = c0, n1 = c1, n2 = c2, n3 = c3;
        if (kc < 7) {
            const float* ap = arow + (kc + 1) * 32;
            n0 = *(const float4*)(ap);
            n1 = *(const float4*)(ap + 4);
            n2 = *(const float4*)(ap + 16);
            n3 = *(const float4*)(ap + 20);
        }

        v16bf a;
        a[0]  = (bf16)c0.x; a[1]  = (bf16)c0.y; a[2]  = (bf16)c0.z; a[3]  = (bf16)c0.w;
        a[4]  = (bf16)c1.x; a[5]  = (bf16)c1.y; a[6]  = (bf16)c1.z; a[7]  = (bf16)c1.w;
        a[8]  = (bf16)c2.x; a[9]  = (bf16)c2.y; a[10] = (bf16)c2.z; a[11] = (bf16)c2.w;
        a[12] = (bf16)c3.x; a[13] = (bf16)c3.y; a[14] = (bf16)c3.z; a[15] = (bf16)c3.w;

        const bf16* bbase = Wl + (size_t)(kc * 8 * 32 + lane) * 16;

        // double-buffered B: issue load of fragment nt+1 before WMMA nt
        v16bf bcur = *(const v16bf*)(bbase);
#pragma unroll
        for (int nt = 0; nt < 8; ++nt) {
            v16bf bnext = bcur;
            if (nt < 7)
                bnext = *(const v16bf*)(bbase + (size_t)(nt + 1) * 32 * 16);
            acc[nt] = __builtin_amdgcn_wmma_f32_16x16x32_bf16(
                false, a, false, bcur, (short)0, acc[nt], false, false);
            bcur = bnext;
        }
        c0 = n0; c1 = n1; c2 = n2; c3 = n3;
    }

    // C layout: VGPR j -> row rowBase + j + 8*khalf, col nt*16 + rlo
#pragma unroll
    for (int nt = 0; nt < 8; ++nt) {
#pragma unroll
        for (int j = 0; j < 8; ++j) {
            int r = rowBase + j + 8 * khalf;
            Y[(size_t)r * 128 + nt * 16 + rlo] = acc[nt][j];
        }
    }
}

// ---------------------------------------------------------------------------
// Node attention: aA/aB[row, h] = sum_d F[row,h,d] * attnA/B[h,d].
// One wave per row; lane covers d-range lane*4..+3, head = lane >> 3.
// ---------------------------------------------------------------------------
__global__ __launch_bounds__(256) void attn_dot2(const float* __restrict__ F,
                                                 const float* __restrict__ attnA,
                                                 const float* __restrict__ attnB,
                                                 float* __restrict__ outA,
                                                 float* __restrict__ outB, int rows) {
    int gid  = blockIdx.x * 256 + threadIdx.x;
    int row  = gid >> 5;
    int lane = gid & 31;
    if (row >= rows) return;
    float4 v  = *(const float4*)(F + (size_t)row * 128 + lane * 4);
    float4 wa = *(const float4*)(attnA + lane * 4);
    float4 wb = *(const float4*)(attnB + lane * 4);
    float pa = v.x * wa.x + v.y * wa.y + v.z * wa.z + v.w * wa.w;
    float pb = v.x * wb.x + v.y * wb.y + v.z * wb.z + v.w * wb.w;
    pa += __shfl_xor(pa, 1, 32); pb += __shfl_xor(pb, 1, 32);
    pa += __shfl_xor(pa, 2, 32); pb += __shfl_xor(pb, 2, 32);
    pa += __shfl_xor(pa, 4, 32); pb += __shfl_xor(pb, 4, 32);
    if ((lane & 7) == 0) {
        outA[(size_t)row * 4 + (lane >> 3)] = pa;
        outB[(size_t)row * 4 + (lane >> 3)] = pb;
    }
}

// Single-dot version (edge features vs attn_e).
__global__ __launch_bounds__(256) void attn_dot(const float* __restrict__ F,
                                                const float* __restrict__ attn,
                                                float* __restrict__ out, int rows) {
    int gid  = blockIdx.x * 256 + threadIdx.x;
    int row  = gid >> 5;
    int lane = gid & 31;
    if (row >= rows) return;
    float4 v = *(const float4*)(F + (size_t)row * 128 + lane * 4);
    float4 w = *(const float4*)(attn + lane * 4);
    float p = v.x * w.x + v.y * w.y + v.z * w.z + v.w * w.w;
    p += __shfl_xor(p, 1, 32);
    p += __shfl_xor(p, 2, 32);
    p += __shfl_xor(p, 4, 32);
    if ((lane & 7) == 0) out[(size_t)row * 4 + (lane >> 3)] = p;
}

__global__ __launch_bounds__(256) void init_buf(float* __restrict__ nodeAcc,
                                                float* __restrict__ m,
                                                float* __restrict__ s,
                                                int n128, int n4) {
    int idx = blockIdx.x * 256 + threadIdx.x;
    if (idx < n128) nodeAcc[idx] = 0.f;
    if (idx < n4) { m[idx] = -__builtin_inff(); s[idx] = 0.f; }
}

__device__ inline void atomicMaxF(float* addr, float val) {
    if (!(val < 0.f)) atomicMax((int*)addr, __float_as_int(val));
    else              atomicMin((unsigned int*)addr, __float_as_uint(val));
}

// logits = leaky_relu(a1[src] + a3 + a2[dst]); store in place over a3;
// segment max into m via atomics.
__global__ __launch_bounds__(256) void logits_max(const float* __restrict__ a1,
                                                  const float* __restrict__ a2,
                                                  float* __restrict__ a3,
                                                  const int* __restrict__ src,
                                                  const int* __restrict__ dst,
                                                  float* __restrict__ m, int E4) {
    int idx = blockIdx.x * 256 + threadIdx.x;
    if (idx >= E4) return;
    int e = idx >> 2, h = idx & 3;
    int d = dst[e];
    float l = a1[(size_t)src[e] * 4 + h] + a3[idx] + a2[(size_t)d * 4 + h];
    l = l > 0.f ? l : ALPHA * l;
    a3[idx] = l;
    atomicMaxF(&m[(size_t)d * 4 + h], l);
}

// ex = exp(logits - m[dst]); store in place; segment sum into s.
__global__ __launch_bounds__(256) void ex_sum(float* __restrict__ lg,
                                              const float* __restrict__ m,
                                              float* __restrict__ s,
                                              const int* __restrict__ dst, int E4) {
    int idx = blockIdx.x * 256 + threadIdx.x;
    if (idx >= E4) return;
    int e = idx >> 2, h = idx & 3;
    int d = dst[e];
    float ex = __expf(lg[idx] - m[(size_t)d * 4 + h]);
    lg[idx] = ex;
    atomicAdd(&s[(size_t)d * 4 + h], ex);
}

// f = (ex/s[dst]) * (e_ft + ft[src]); write elu(f) in place over e_ft
// (which lives in the ret2 region of d_out); scatter-add f into nodeAcc.
__global__ __launch_bounds__(256) void edge_out(float* __restrict__ eft,
                                                const float* __restrict__ ft,
                                                const float* __restrict__ ex,
                                                const float* __restrict__ s,
                                                const int* __restrict__ src,
                                                const int* __restrict__ dst,
                                                float* __restrict__ nodeAcc, int E) {
    int gid  = blockIdx.x * 256 + threadIdx.x;
    int e    = gid >> 5;
    int lane = gid & 31;
    if (e >= E) return;
    int h  = lane >> 3;
    int dn = dst[e], sn = src[e];
    float a = ex[(size_t)e * 4 + h] / s[(size_t)dn * 4 + h];
    size_t eo = (size_t)e * 128 + lane * 4;
    float4 ef = *(const float4*)(eft + eo);
    float4 ff = *(const float4*)(ft + (size_t)sn * 128 + lane * 4);
    float4 f;
    f.x = a * (ef.x + ff.x); f.y = a * (ef.y + ff.y);
    f.z = a * (ef.z + ff.z); f.w = a * (ef.w + ff.w);
    float4 o;
    o.x = f.x > 0.f ? f.x : __expf(f.x) - 1.f;
    o.y = f.y > 0.f ? f.y : __expf(f.y) - 1.f;
    o.z = f.z > 0.f ? f.z : __expf(f.z) - 1.f;
    o.w = f.w > 0.f ? f.w : __expf(f.w) - 1.f;
    *(float4*)(eft + eo) = o;
    float* np = nodeAcc + (size_t)dn * 128 + lane * 4;
    atomicAdd(np + 0, f.x);
    atomicAdd(np + 1, f.y);
    atomicAdd(np + 2, f.z);
    atomicAdd(np + 3, f.w);
}

__global__ __launch_bounds__(256) void elu_out(const float* __restrict__ in,
                                               float* __restrict__ out, int n) {
    int idx = blockIdx.x * 256 + threadIdx.x;
    if (idx >= n) return;
    float v = in[idx];
    out[idx] = v > 0.f ? v : __expf(v) - 1.f;
}

// ---------------------------------------------------------------------------
extern "C" void kernel_launch(void* const* d_in, const int* in_sizes, int n_in,
                              void* d_out, int out_size, void* d_ws, size_t ws_size,
                              hipStream_t stream) {
    const float* nodef  = (const float*)d_in[0];
    const float* edgef  = (const float*)d_in[1];
    const int*   src    = (const int*)d_in[2];
    const int*   dst    = (const int*)d_in[3];
    const float* W      = (const float*)d_in[4];
    const float* attn_l = (const float*)d_in[5];
    const float* attn_r = (const float*)d_in[6];
    const float* attn_e = (const float*)d_in[7];
    (void)n_in; (void)out_size; (void)ws_size;

    const int N = in_sizes[0] / 256;
    const int E = in_sizes[1] / 256;

    float* ret1 = (float*)d_out;                 // N*128
    float* eft  = ret1 + (size_t)N * 128;        // E*128: e_ft, then elu(f_cat)

    char* ws = (char*)d_ws;
    size_t off = 0;
    auto take = [&](size_t bytes) -> char* {
        char* p = ws + off;
        off = (off + bytes + 255) & ~(size_t)255;
        return p;
    };
    unsigned short* Wp   = (unsigned short*)take(65536);
    float* ft      = (float*)take((size_t)N * 128 * 4);
    float* nodeAcc = (float*)take((size_t)N * 128 * 4);
    float* a1      = (float*)take((size_t)N * 16);
    float* a2      = (float*)take((size_t)N * 16);
    float* m       = (float*)take((size_t)N * 16);
    float* s       = (float*)take((size_t)N * 16);
    float* a3      = (float*)take((size_t)E * 16);   // a3 -> logits -> ex

    pack_w<<<256, 256, 0, stream>>>(W, Wp);

    gat_gemm<<<(N + 127) / 128, 256, 0, stream>>>(nodef, Wp, ft, N);
    gat_gemm<<<(E + 127) / 128, 256, 0, stream>>>(edgef, Wp, eft, E);

    int n128 = N * 128, n4 = N * 4, e4 = E * 4;
    init_buf<<<(n128 + 255) / 256, 256, 0, stream>>>(nodeAcc, m, s, n128, n4);

    attn_dot2<<<((size_t)N * 32 + 255) / 256, 256, 0, stream>>>(ft, attn_l, attn_r,
                                                                a1, a2, N);
    attn_dot<<<((size_t)E * 32 + 255) / 256, 256, 0, stream>>>(eft, attn_e, a3, E);

    logits_max<<<(e4 + 255) / 256, 256, 0, stream>>>(a1, a2, a3, src, dst, m, e4);
    ex_sum<<<(e4 + 255) / 256, 256, 0, stream>>>(a3, m, s, dst, e4);

    edge_out<<<((size_t)E * 32 + 255) / 256, 256, 0, stream>>>(eft, ft, a3, s, src, dst,
                                                               nodeAcc, E);
    elu_out<<<(n128 + 255) / 256, 256, 0, stream>>>(nodeAcc, ret1, n128);
}
